// BidirectionalAttention_35467839930684
// MI455X (gfx1250) — matrix operand order repaired and checked by reading, not compiled, not run on hardware
//
#include <hip/hip_runtime.h>

// ---------------------------------------------------------------------------
// BidirectionalAttention for MI455X (gfx1250), wave32 + WMMA bf16.
// B=2, L=2048, D=1024, H=16, HD=64.
// Round 2: doubled register-level reuse (2 M-tiles / 2 Q-tiles per wave).
// ---------------------------------------------------------------------------

#define DEV static __device__ __forceinline__

typedef __bf16 bf16_t;
typedef __attribute__((ext_vector_type(16))) __bf16 bf16x16;
typedef __attribute__((ext_vector_type(8)))  __bf16 bf16x8;
typedef __attribute__((ext_vector_type(8)))  float  f32x8;
typedef __attribute__((ext_vector_type(4)))  float  f32x4;
typedef __attribute__((ext_vector_type(8)))  unsigned int   u32x8;
typedef __attribute__((ext_vector_type(8)))  unsigned short u16x8;
typedef __attribute__((ext_vector_type(4)))  unsigned short u16x4;

#define NB   2
#define NL   2048
#define ND   1024
#define NH   16
#define NHD  64

// float -> bf16 (round to nearest even-ish)
DEV unsigned short f2bf(float f) {
  unsigned int u = __builtin_bit_cast(unsigned int, f);
  u += 0x7FFFu + ((u >> 16) & 1u);
  return (unsigned short)(u >> 16);
}

// A-fragment (16x32 bf16): lane holds row M=lane&15; K elems
// {2kb..2kb+15} with kb = (lane>=16 ? 8 : 0). Two 16B loads.
DEV bf16x16 ld_a32(const bf16_t* p, int kb) {
  bf16x8 lo = *reinterpret_cast<const bf16x8*>(p + 2 * kb);
  bf16x8 hi = *reinterpret_cast<const bf16x8*>(p + 2 * kb + 8);
  bf16x16 r;
#pragma unroll
  for (int i = 0; i < 8; ++i) { r[i] = lo[i]; r[i + 8] = hi[i]; }
  return r;
}

DEV f32x8 wmma_bf(bf16x16 a, bf16x16 b, f32x8 c) {
  return __builtin_amdgcn_wmma_f32_16x16x32_bf16(false, a, false, b,
                                                 (short)0, c, false, false);
}

// ---------------------------------------------------------------------------
// Conversion kernels (4-wide)
// ---------------------------------------------------------------------------
__global__ __launch_bounds__(256) void cvt_bf16_kernel(const float* __restrict__ s,
                                                       unsigned short* __restrict__ d,
                                                       int n4) {
  for (int i = blockIdx.x * blockDim.x + threadIdx.x; i < n4;
       i += gridDim.x * blockDim.x) {
    f32x4 v = reinterpret_cast<const f32x4*>(s)[i];
    u16x4 o;
#pragma unroll
    for (int j = 0; j < 4; ++j) o[j] = f2bf(v[j]);
    reinterpret_cast<u16x4*>(d)[i] = o;
  }
}

__global__ __launch_bounds__(256) void mask_bias_kernel(const unsigned char* __restrict__ m,
                                                        float* __restrict__ d, int n) {
  for (int i = blockIdx.x * blockDim.x + threadIdx.x; i < n;
       i += gridDim.x * blockDim.x)
    d[i] = m[i] ? -1.0e30f : 0.0f;
}

// ---------------------------------------------------------------------------
// QKV GEMM: qkv[m][n] = sum_k x[m][k] * wqkv[n][k] + bqkv[n]
// M = B*L = 4096, N = 3*D = 3072, K = D = 1024.
// One wave -> 32 x 64 output block (2 m-tiles reuse 4 B-fragments).
// Output routed into Q/K (b,h,l,d) and V transposed Vt (b,h,d,l), bf16.
// ---------------------------------------------------------------------------
__global__ __launch_bounds__(256) void qkv_gemm_kernel(
    const unsigned short* __restrict__ xb_,  const unsigned short* __restrict__ wb_,
    const float* __restrict__ bias,
    unsigned short* __restrict__ Qm, unsigned short* __restrict__ Km,
    unsigned short* __restrict__ Vt) {
  const bf16_t* xb = reinterpret_cast<const bf16_t*>(xb_);
  const bf16_t* wb = reinterpret_cast<const bf16_t*>(wb_);

  int w    = (blockIdx.x * blockDim.x + threadIdx.x) >> 5;
  int lane = threadIdx.x & 31;
  int half = lane >> 4;
  int qn   = lane & 15;
  int nt   = w % 48;          // 48 strips of 64 over N=3072
  int mt   = w / 48;          // 128 blocks of 32 rows over M=4096
  int kb16 = half * 16, kb8 = half * 8;

  const bf16_t* a0row = xb + (size_t)(mt * 32 + qn) * ND;
  const bf16_t* a1row = a0row + (size_t)16 * ND;
  const bf16_t* w0 = wb + (size_t)(nt * 64 +  0 + qn) * ND;
  const bf16_t* w1 = wb + (size_t)(nt * 64 + 16 + qn) * ND;
  const bf16_t* w2 = wb + (size_t)(nt * 64 + 32 + qn) * ND;
  const bf16_t* w3 = wb + (size_t)(nt * 64 + 48 + qn) * ND;

  f32x8 z = {0.f,0.f,0.f,0.f,0.f,0.f,0.f,0.f};
  f32x8 c00 = z, c01 = z, c02 = z, c03 = z;
  f32x8 c10 = z, c11 = z, c12 = z, c13 = z;

  for (int k = 0; k < ND; k += 32) {
    bf16x16 a0 = ld_a32(a0row + k, kb8);
    bf16x16 a1 = ld_a32(a1row + k, kb8);
    bf16x16 b0 = *reinterpret_cast<const bf16x16*>(w0 + k + kb16);
    bf16x16 b1 = *reinterpret_cast<const bf16x16*>(w1 + k + kb16);
    bf16x16 b2 = *reinterpret_cast<const bf16x16*>(w2 + k + kb16);
    bf16x16 b3 = *reinterpret_cast<const bf16x16*>(w3 + k + kb16);
    c00 = wmma_bf(a0, b0, c00);  c10 = wmma_bf(a1, b0, c10);
    c01 = wmma_bf(a0, b1, c01);  c11 = wmma_bf(a1, b1, c11);
    c02 = wmma_bf(a0, b2, c02);  c12 = wmma_bf(a1, b2, c12);
    c03 = wmma_bf(a0, b3, c03);  c13 = wmma_bf(a1, b3, c13);
  }

  int nb = nt * 64;
  int s  = nb >> 10;               // 0=Q 1=K 2=V
  int hh = (nb & 1023) >> 6;       // head
  float bj0 = bias[nb +  0 + qn];
  float bj1 = bias[nb + 16 + qn];
  float bj2 = bias[nb + 32 + qn];
  float bj3 = bias[nb + 48 + qn];

  auto store_tile = [&](f32x8 c0, f32x8 c1, f32x8 c2, f32x8 c3, int rbase) {
    int b  = rbase >> 11;            // /2048
    int l0 = rbase & 2047;
    if (s < 2) {
      unsigned short* dst = (s == 0 ? Qm : Km) + ((size_t)(b * NH + hh) * NL) * NHD;
#pragma unroll
      for (int r = 0; r < 8; ++r) {
        dst[(size_t)(l0 + r) * NHD +  0 + qn] = f2bf(c0[r] + bj0);
        dst[(size_t)(l0 + r) * NHD + 16 + qn] = f2bf(c1[r] + bj1);
        dst[(size_t)(l0 + r) * NHD + 32 + qn] = f2bf(c2[r] + bj2);
        dst[(size_t)(l0 + r) * NHD + 48 + qn] = f2bf(c3[r] + bj3);
      }
    } else {
      unsigned short* dbase = Vt + ((size_t)(b * NH + hh) * NHD) * NL + l0;
      u16x8 p0, p1, p2, p3;
#pragma unroll
      for (int r = 0; r < 8; ++r) {
        p0[r] = f2bf(c0[r] + bj0); p1[r] = f2bf(c1[r] + bj1);
        p2[r] = f2bf(c2[r] + bj2); p3[r] = f2bf(c3[r] + bj3);
      }
      *reinterpret_cast<u16x8*>(dbase + (size_t)( 0 + qn) * NL) = p0;
      *reinterpret_cast<u16x8*>(dbase + (size_t)(16 + qn) * NL) = p1;
      *reinterpret_cast<u16x8*>(dbase + (size_t)(32 + qn) * NL) = p2;
      *reinterpret_cast<u16x8*>(dbase + (size_t)(48 + qn) * NL) = p3;
    }
  };
  store_tile(c00, c01, c02, c03, mt * 32 +      half * 8);
  store_tile(c10, c11, c12, c13, mt * 32 + 16 + half * 8);
}

// ---------------------------------------------------------------------------
// Flash attention, transposed formulation. One wave = (b, h, 32-query block).
// S^T = K * Q^T  ->  lane owns one query column; softmax stats per lane.
// K and V A-fragments are shared by both 16-query tiles (2x intensity).
// P^T C-frag -> B-frag via one xor-16 register exchange (no LDS round trip).
// ---------------------------------------------------------------------------
__global__ __launch_bounds__(256) void attn_kernel(
    const unsigned short* __restrict__ Qm_, const unsigned short* __restrict__ Km_,
    const unsigned short* __restrict__ Vt_, const float* __restrict__ mbias,
    unsigned short* __restrict__ Obuf) {
  const bf16_t* Qm = reinterpret_cast<const bf16_t*>(Qm_);
  const bf16_t* Km = reinterpret_cast<const bf16_t*>(Km_);
  const bf16_t* Vt = reinterpret_cast<const bf16_t*>(Vt_);

  int w    = (blockIdx.x * blockDim.x + threadIdx.x) >> 5;
  int lane = threadIdx.x & 31;
  int half = lane >> 4;
  int qn   = lane & 15;
  int qt   = w & 63;           // 64 blocks of 32 queries
  int bh   = w >> 6;           // b*16 + h
  int kb8  = half * 8;

  // Q B-fragments (loop invariant), one pair per 16-query tile.
  bf16x16 bq0[2], bq1[2];
#pragma unroll
  for (int t = 0; t < 2; ++t) {
    const bf16_t* qrow = Qm + ((size_t)bh * NL + qt * 32 + t * 16 + qn) * NHD;
    bq0[t] = *reinterpret_cast<const bf16x16*>(qrow + half * 16);
    bq1[t] = *reinterpret_cast<const bf16x16*>(qrow + 32 + half * 16);
  }

  const bf16_t* Kb = Km + (size_t)bh * NL * NHD;
  const bf16_t* Vb = Vt + (size_t)bh * NHD * NL;
  const float*  mb = mbias + (size_t)(bh >> 4) * NL;

  f32x8 z = {0.f,0.f,0.f,0.f,0.f,0.f,0.f,0.f};
  f32x8 o[2][4];
#pragma unroll
  for (int t = 0; t < 2; ++t)
#pragma unroll
    for (int d = 0; d < 4; ++d) o[t][d] = z;
  float mrun[2] = {-3.0e38f, -3.0e38f};
  float lrun[2] = {0.0f, 0.0f};
  const float scale = 0.125f;  // 1/sqrt(64)

  for (int jb = 0; jb < NL; jb += 32) {
    // Shared K A-fragments for this 32-key block.
    const bf16_t* k0 = Kb + (size_t)(jb + qn) * NHD;
    const bf16_t* k1 = k0 + 16 * NHD;
    bf16x16 a00 = ld_a32(k0,      kb8);
    bf16x16 a01 = ld_a32(k0 + 32, kb8);
    bf16x16 a10 = ld_a32(k1,      kb8);
    bf16x16 a11 = ld_a32(k1 + 32, kb8);

    // Padding-mask bias for this lane's key rows (same for both q-tiles).
    f32x4 m0 = *reinterpret_cast<const f32x4*>(mb + jb + kb8);
    f32x4 m1 = *reinterpret_cast<const f32x4*>(mb + jb + kb8 + 4);
    f32x4 m2 = *reinterpret_cast<const f32x4*>(mb + jb + kb8 + 16);
    f32x4 m3 = *reinterpret_cast<const f32x4*>(mb + jb + kb8 + 20);

    bf16x16 bp[2];
#pragma unroll
    for (int t = 0; t < 2; ++t) {
      f32x8 c0 = wmma_bf(a00, bq0[t], z);
      c0       = wmma_bf(a01, bq1[t], c0);
      f32x8 c1 = wmma_bf(a10, bq0[t], z);
      c1       = wmma_bf(a11, bq1[t], c1);

      float s0[8], s1[8];
#pragma unroll
      for (int r = 0; r < 8; ++r) {
        s0[r] = c0[r] * scale + (r < 4 ? m0[r] : m1[r - 4]);
        s1[r] = c1[r] * scale + (r < 4 ? m2[r] : m3[r - 4]);
      }
      float tm = s0[0];
#pragma unroll
      for (int r = 0; r < 8; ++r) { tm = fmaxf(tm, s0[r]); tm = fmaxf(tm, s1[r]); }
      tm = fmaxf(tm, __shfl_xor(tm, 16, 32));
      float mnew  = fmaxf(mrun[t], tm);
      float alpha = __expf(mrun[t] - mnew);

      float p0[8], p1[8], rs = 0.f;
#pragma unroll
      for (int r = 0; r < 8; ++r) {
        p0[r] = __expf(s0[r] - mnew); rs += p0[r];
        p1[r] = __expf(s1[r] - mnew); rs += p1[r];
      }
      rs += __shfl_xor(rs, 16, 32);
      lrun[t] = lrun[t] * alpha + rs;
      mrun[t] = mnew;
#pragma unroll
      for (int r = 0; r < 8; ++r) {
        o[t][0][r] *= alpha; o[t][1][r] *= alpha;
        o[t][2][r] *= alpha; o[t][3][r] *= alpha;
      }

      // P^T C-fragment -> B-fragment: pack bf16 pairs, exchange halves.
      unsigned int pk0[4], pk1[4], rcv[4];
#pragma unroll
      for (int i = 0; i < 4; ++i) {
        pk0[i] = (unsigned)f2bf(p0[2*i]) | ((unsigned)f2bf(p0[2*i+1]) << 16);
        pk1[i] = (unsigned)f2bf(p1[2*i]) | ((unsigned)f2bf(p1[2*i+1]) << 16);
      }
#pragma unroll
      for (int i = 0; i < 4; ++i) {
        unsigned int snd = half ? pk0[i] : pk1[i];
        rcv[i] = __shfl_xor(snd, 16, 32);
      }
      u32x8 bpu;
#pragma unroll
      for (int i = 0; i < 4; ++i) {
        bpu[i]     = half ? rcv[i] : pk0[i];
        bpu[4 + i] = half ? pk1[i] : rcv[i];
      }
      bp[t] = __builtin_bit_cast(bf16x16, bpu);
    }

    // O^T += V^T * P^T : V A-fragments shared by both q-tiles.
#pragma unroll
    for (int dt = 0; dt < 4; ++dt) {
      bf16x16 av = ld_a32(Vb + (size_t)(dt * 16 + qn) * NL + jb, kb8);
      o[0][dt] = wmma_bf(av, bp[0], o[0][dt]);
      o[1][dt] = wmma_bf(av, bp[1], o[1][dt]);
    }
  }

  // Store O as (b, l, h*64+d) bf16: lane owns one l (q), d contiguous -> 16B.
#pragma unroll
  for (int t = 0; t < 2; ++t) {
    float linv = 1.0f / lrun[t];
    unsigned short* orow =
        Obuf + ((size_t)(bh >> 4) * NL + qt * 32 + t * 16 + qn) * ND +
        (bh & 15) * NHD + kb8;
#pragma unroll
    for (int dt = 0; dt < 4; ++dt) {
      u16x8 sv;
#pragma unroll
      for (int r = 0; r < 8; ++r) sv[r] = f2bf(o[t][dt][r] * linv);
      *reinterpret_cast<u16x8*>(orow + dt * 16) = sv;
    }
  }
}

// ---------------------------------------------------------------------------
// Output projection: out[m][n] = sum_k O[m][k] * wproj[n][k] + bproj[n], f32.
// M = 4096, N = K = 1024.  One wave -> 32 x 64 output block.
// ---------------------------------------------------------------------------
__global__ __launch_bounds__(256) void proj_gemm_kernel(
    const unsigned short* __restrict__ ob_, const unsigned short* __restrict__ wb_,
    const float* __restrict__ bias, float* __restrict__ out) {
  const bf16_t* ob = reinterpret_cast<const bf16_t*>(ob_);
  const bf16_t* wb = reinterpret_cast<const bf16_t*>(wb_);

  int w    = (blockIdx.x * blockDim.x + threadIdx.x) >> 5;
  int lane = threadIdx.x & 31;
  int half = lane >> 4;
  int qn   = lane & 15;
  int nt   = w & 15;           // 16 strips of 64 over N=1024
  int mt   = w >> 4;           // 128 blocks of 32 rows
  int kb16 = half * 16, kb8 = half * 8;

  const bf16_t* a0row = ob + (size_t)(mt * 32 + qn) * ND;
  const bf16_t* a1row = a0row + (size_t)16 * ND;
  const bf16_t* w0 = wb + (size_t)(nt * 64 +  0 + qn) * ND;
  const bf16_t* w1 = wb + (size_t)(nt * 64 + 16 + qn) * ND;
  const bf16_t* w2 = wb + (size_t)(nt * 64 + 32 + qn) * ND;
  const bf16_t* w3 = wb + (size_t)(nt * 64 + 48 + qn) * ND;

  f32x8 z = {0.f,0.f,0.f,0.f,0.f,0.f,0.f,0.f};
  f32x8 c00 = z, c01 = z, c02 = z, c03 = z;
  f32x8 c10 = z, c11 = z, c12 = z, c13 = z;

  for (int k = 0; k < ND; k += 32) {
    bf16x16 a0 = ld_a32(a0row + k, kb8);
    bf16x16 a1 = ld_a32(a1row + k, kb8);
    bf16x16 b0 = *reinterpret_cast<const bf16x16*>(w0 + k + kb16);
    bf16x16 b1 = *reinterpret_cast<const bf16x16*>(w1 + k + kb16);
    bf16x16 b2 = *reinterpret_cast<const bf16x16*>(w2 + k + kb16);
    bf16x16 b3 = *reinterpret_cast<const bf16x16*>(w3 + k + kb16);
    c00 = wmma_bf(a0, b0, c00);  c10 = wmma_bf(a1, b0, c10);
    c01 = wmma_bf(a0, b1, c01);  c11 = wmma_bf(a1, b1, c11);
    c02 = wmma_bf(a0, b2, c02);  c12 = wmma_bf(a1, b2, c12);
    c03 = wmma_bf(a0, b3, c03);  c13 = wmma_bf(a1, b3, c13);
  }

  int nb  = nt * 64;
  float b0 = bias[nb +  0 + qn], b1 = bias[nb + 16 + qn];
  float b2 = bias[nb + 32 + qn], b3 = bias[nb + 48 + qn];

  auto store_tile = [&](f32x8 c0, f32x8 c1, f32x8 c2, f32x8 c3, int rbase) {
#pragma unroll
    for (int r = 0; r < 8; ++r) {
      float* orow = out + (size_t)(rbase + r) * ND + nb;
      orow[ 0 + qn] = c0[r] + b0;
      orow[16 + qn] = c1[r] + b1;
      orow[32 + qn] = c2[r] + b2;
      orow[48 + qn] = c3[r] + b3;
    }
  };
  store_tile(c00, c01, c02, c03, mt * 32 +      half * 8);
  store_tile(c10, c11, c12, c13, mt * 32 + 16 + half * 8);
}

// ---------------------------------------------------------------------------
// Launch
// ---------------------------------------------------------------------------
extern "C" void kernel_launch(void* const* d_in, const int* in_sizes, int n_in,
                              void* d_out, int out_size, void* d_ws, size_t ws_size,
                              hipStream_t stream) {
  const float*         x     = (const float*)d_in[0];
  const unsigned char* mask  = (const unsigned char*)d_in[1];  // jax bool -> 1 byte
  const float*         wqkv  = (const float*)d_in[2];
  const float*         bqkv  = (const float*)d_in[3];
  const float*         wproj = (const float*)d_in[4];
  const float*         bproj = (const float*)d_in[5];
  float*               out   = (float*)d_out;

  const size_t nX  = (size_t)NB * NL * ND;       // 4,194,304
  const size_t nWQ = (size_t)3 * ND * ND;        // 3,145,728
  const size_t nWP = (size_t)ND * ND;            // 1,048,576
  const size_t nBL = (size_t)NB * NL;            // 4,096

  char* ws = (char*)d_ws;
  size_t off = 0;
  auto carve = [&](size_t bytes) { char* p = ws + off; off += (bytes + 255) & ~(size_t)255; return p; };

  unsigned short* xb     = (unsigned short*)carve(nX  * 2);
  unsigned short* wqkvb  = (unsigned short*)carve(nWQ * 2);
  unsigned short* wprojb = (unsigned short*)carve(nWP * 2);
  unsigned short* Qm     = (unsigned short*)carve(nX  * 2);
  unsigned short* Km     = (unsigned short*)carve(nX  * 2);
  unsigned short* Vt     = (unsigned short*)carve(nX  * 2);
  unsigned short* Obuf   = (unsigned short*)carve(nX  * 2);
  float*          mbias  = (float*)carve(nBL * 4);
  (void)ws_size; (void)in_sizes; (void)n_in; (void)out_size;

  cvt_bf16_kernel<<<512, 256, 0, stream>>>(x,     xb,     (int)(nX  / 4));
  cvt_bf16_kernel<<<512, 256, 0, stream>>>(wqkv,  wqkvb,  (int)(nWQ / 4));
  cvt_bf16_kernel<<<256, 256, 0, stream>>>(wproj, wprojb, (int)(nWP / 4));
  mask_bias_kernel<<<16, 256, 0, stream>>>(mask, mbias, (int)nBL);

  // QKV GEMM: 128 m-blocks * 48 n-strips = 6144 waves, 8 waves/block
  qkv_gemm_kernel<<<768, 256, 0, stream>>>(xb, wqkvb, bqkv, Qm, Km, Vt);
  // Attention: B*H*(L/32) = 2048 waves
  attn_kernel<<<256, 256, 0, stream>>>(Qm, Km, Vt, mbias, Obuf);
  // Projection: 128 * 16 = 2048 waves
  proj_gemm_kernel<<<256, 256, 0, stream>>>(Obuf, wprojb, bproj, out);
}
